// RelationPredictor_18141941858697
// MI455X (gfx1250) — matrix operand order repaired and checked by reading, not compiled, not run on hardware
//
#include <hip/hip_runtime.h>
#include <hip/hip_bf16.h>

typedef __attribute__((ext_vector_type(16))) _Float16 v16h;
typedef __attribute__((ext_vector_type(8)))  float    v8f;

#define BB 8
#define NN 256
#define DD 256
#define HH 128
#define RR 8

// ---------------------------------------------------------------------------
// Kernel 1: Pi[row,h] = slots[row,:] . W1[0:256, h] + b1[h]   (b1 folded in)
//           Pj[row,h] = slots[row,:] . W1[256:512, h]
// One block per slot row (B*N = 2048 rows), 128 threads (one per h).
// Slot row staged in LDS; W1 reads coalesced across threads.
// ---------------------------------------------------------------------------
__global__ void proj_kernel(const float* __restrict__ slots,
                            const float* __restrict__ W1,
                            const float* __restrict__ b1,
                            float* __restrict__ Pi,
                            float* __restrict__ Pj) {
    __shared__ float s[DD];
    const int row = blockIdx.x;
    const int h   = threadIdx.x;                 // 0..127
    for (int d = h; d < DD; d += HH) s[d] = slots[row * DD + d];
    __syncthreads();
    float accI = b1[h];
    float accJ = 0.0f;
#pragma unroll 8
    for (int d = 0; d < DD; ++d) {
        const float sv = s[d];
        accI = fmaf(sv, W1[d * HH + h], accI);
        accJ = fmaf(sv, W1[(DD + d) * HH + h], accJ);
    }
    Pi[row * HH + h] = accI;
    Pj[row * HH + h] = accJ;
}

// ---------------------------------------------------------------------------
// Kernel 2: one wave per 16-pair tile (fixed b,i ; j in [j0, j0+16)).
//   A[m,k] = relu(Pi[b,i,k] + Pj[b,j0+m,k])  built in registers as f16
//   B[k,r] = W2[k,r] (r<8), 0-padded to 16 cols
//   D = A x B accumulated over K=128 via 4x v_wmma_f32_16x16x32_f16
//   out = sigmoid(D + b2)
// ---------------------------------------------------------------------------
__global__ void pair_kernel(const float* __restrict__ Pi,
                            const float* __restrict__ Pj,
                            const float* __restrict__ W2,
                            const float* __restrict__ b2,
                            float* __restrict__ out) {
    const int lane = threadIdx.x & 31;
    const int wave = threadIdx.x >> 5;
    const int w    = blockIdx.x * 8 + wave;      // global tile id, 0..32767

    const int tilesPerB = NN * (NN / 16);        // 4096
    const int b   = w / tilesPerB;
    const int rem = w % tilesPerB;
    const int i   = rem / (NN / 16);
    const int j0  = (rem % (NN / 16)) * 16;

    const int n   = lane & 15;                   // column (r) / A-row m
    const int sel = lane >> 4;                   // K-half selector

    // --- B fragments: W2 padded to 16 columns, per ISA 16-bit B 32x16 layout:
    // lane L holds column n=L%16; K = sel*8 + t (t<8), 16 + sel*8 + (t-8) (t>=8) within each 32-K chunk.
    v16h bf[4];
#pragma unroll
    for (int c = 0; c < 4; ++c) {
#pragma unroll
        for (int t = 0; t < 16; ++t) {
            const int k = c * 32 + sel * 8 + (t < 8 ? t : t + 8);
            const float wv = (n < RR) ? W2[k * RR + n] : 0.0f;
            bf[c][t] = (_Float16)wv;
        }
    }

    const float* __restrict__ piRow = Pi + (b * NN + i) * HH;
    const float* __restrict__ pjRow = Pj + (b * NN + (j0 + n)) * HH; // row m = lane%16

    v8f acc = {};
#pragma unroll
    for (int c = 0; c < 4; ++c) {
        // ISA 16-bit A 16x32 layout: lane holds K = {kc+sel*8 .. +7} and
        // {kc+16+sel*8 .. +7} for row m = lane%16.
        const int k1 = c * 32 + sel * 8;
        const int k2 = k1 + 16;

        float pi_v[16], pj_v[16];
        *(float4*)(pi_v + 0)  = *(const float4*)(piRow + k1);
        *(float4*)(pi_v + 4)  = *(const float4*)(piRow + k1 + 4);
        *(float4*)(pi_v + 8)  = *(const float4*)(piRow + k2);
        *(float4*)(pi_v + 12) = *(const float4*)(piRow + k2 + 4);
        *(float4*)(pj_v + 0)  = *(const float4*)(pjRow + k1);
        *(float4*)(pj_v + 4)  = *(const float4*)(pjRow + k1 + 4);
        *(float4*)(pj_v + 8)  = *(const float4*)(pjRow + k2);
        *(float4*)(pj_v + 12) = *(const float4*)(pjRow + k2 + 4);

        v16h a;
#pragma unroll
        for (int t = 0; t < 16; ++t) {
            a[t] = (_Float16)fmaxf(pi_v[t] + pj_v[t], 0.0f);
        }

        acc = __builtin_amdgcn_wmma_f32_16x16x32_f16(
            /*neg_a=*/false, a, /*neg_b=*/false, bf[c],
            /*c_mod=*/(short)0, acc, /*reuse_a=*/false, /*reuse_b=*/false);
    }

    // --- epilogue: bias + sigmoid, store only the 8 valid columns.
    // C/D layout: VGPR v holds row m = v + 8*sel, column n = lane%16.
    if (n < RR) {
        const float bias = b2[n];
#pragma unroll
        for (int v = 0; v < 8; ++v) {
            const int j = j0 + v + 8 * sel;
            const float x = acc[v] + bias;
            const float sg = 1.0f / (1.0f + __expf(-x));
            out[(((b * NN) + i) * NN + j) * RR + n] = sg;
        }
    }
}

extern "C" void kernel_launch(void* const* d_in, const int* in_sizes, int n_in,
                              void* d_out, int out_size, void* d_ws, size_t ws_size,
                              hipStream_t stream) {
    (void)in_sizes; (void)n_in; (void)out_size; (void)ws_size;
    const float* slots = (const float*)d_in[0];   // [8,256,256]
    const float* W1    = (const float*)d_in[1];   // [512,128]
    const float* b1    = (const float*)d_in[2];   // [128]
    const float* W2    = (const float*)d_in[3];   // [128,8]
    const float* b2    = (const float*)d_in[4];   // [8]
    float* out = (float*)d_out;                   // [8,256,256,8]

    float* Pi = (float*)d_ws;                     // [2048,128]
    float* Pj = Pi + BB * NN * HH;                // [2048,128]

    proj_kernel<<<BB * NN, HH, 0, stream>>>(slots, W1, b1, Pi, Pj);

    const int tiles  = BB * NN * (NN / 16);       // 32768 wave-tiles
    const int blocks = tiles / 8;                 // 8 waves per 256-thread block
    pair_kernel<<<blocks, 256, 0, stream>>>(Pi, Pj, W2, b2, out);
}
